// Colored_Net_30709016167062
// MI455X (gfx1250) — matrix operand order repaired and checked by reading, not compiled, not run on hardware
//
#include <hip/hip_runtime.h>
#include <math.h>

typedef __attribute__((ext_vector_type(16))) _Float16 v16h;
typedef __attribute__((ext_vector_type(8)))  float    v8f;

#define N_NODES_REF 100000

// ---------------------------------------------------------------------------
// K1: layer-1 scalar edge aggregation: agg1[dst] += feat[src]
// ---------------------------------------------------------------------------
__global__ void edge_pass1(const int* __restrict__ src, const int* __restrict__ dst,
                           const float* __restrict__ feat, float* __restrict__ agg1,
                           int nEdges) {
    int e = blockIdx.x * blockDim.x + threadIdx.x;
    if (e < nEdges) {
        atomicAdd(&agg1[dst[e]], feat[src[e]]);
    }
}

// ---------------------------------------------------------------------------
// K2: per-node transform via WMMA.
//   h1[i,c]  = relu(agg1[i]*W1_rel[c] + feat[i]*W1_root[c] + b1[c])   (c=0..63)
//   out[i,n] = sum_c h1[i,c] * Wc[n,c], Wc = [W2_rel ; W2_root] (64x64)
//   cols 0..31 -> z1 (to be edge-aggregated), cols 32..63 -> r1 (root path)
// One wave handles a 16-node tile: A = h1 tile (16x64 as two 16x32 f16 chunks),
// B = Wc^T chunks, 8x v_wmma_f32_16x16x32_f16 per tile.
// ---------------------------------------------------------------------------
__global__ void node_transform(const float* __restrict__ feat,
                               const float* __restrict__ agg1,
                               const float* __restrict__ W1_rel,
                               const float* __restrict__ W1_root,
                               const float* __restrict__ b1,
                               const float* __restrict__ W2_rel,
                               const float* __restrict__ W2_root,
                               float* __restrict__ z1,
                               float* __restrict__ r1,
                               int nTiles) {
    int wave = (int)((blockIdx.x * blockDim.x + threadIdx.x) >> 5);
    int lane = threadIdx.x & 31;
    if (wave >= nTiles) return;              // wave-uniform branch; EXEC stays full

    int m    = lane & 15;                    // A-row / B-column / D-column index
    int half = lane >> 4;

    // --- B fragments: B[k][n] = Wc[n][k]; frag(nt,kc): n = nt*16+m, k = kc*32 + half*16 + idx
    v16h Bf[8];
#pragma unroll
    for (int nt = 0; nt < 4; ++nt) {
        int n = nt * 16 + m;
        const float* Wrow = (n < 32) ? (W2_rel + n * 64) : (W2_root + (n - 32) * 64);
#pragma unroll
        for (int kc = 0; kc < 2; ++kc) {
            v16h b;
#pragma unroll
            for (int idx = 0; idx < 16; ++idx) {
                int k = kc * 32 + half * 16 + idx;
                b[idx] = (_Float16)Wrow[k];
            }
            Bf[nt * 2 + kc] = b;
        }
    }

    // --- A fragments: h1 for row m of this tile, k = kc*32 + half*16 + idx (matches B)
    int node = wave * 16 + m;
    float av = agg1[node];
    float fv = feat[node];
    v16h A[2];
#pragma unroll
    for (int kc = 0; kc < 2; ++kc) {
        v16h a;
#pragma unroll
        for (int idx = 0; idx < 16; ++idx) {
            int k = kc * 32 + half * 16 + idx;
            float h = fmaf(av, W1_rel[k], fmaf(fv, W1_root[k], b1[k]));
            a[idx] = (_Float16)(h > 0.f ? h : 0.f);
        }
        A[kc] = a;
    }

    // --- 4 output tiles x 2 K-chunks = 8 WMMAs
    v8f acc[4];
#pragma unroll
    for (int nt = 0; nt < 4; ++nt) {
        v8f c = {};
        c = __builtin_amdgcn_wmma_f32_16x16x32_f16(false, A[0], false, Bf[nt * 2 + 0],
                                                   (short)0, c, false, false);
        c = __builtin_amdgcn_wmma_f32_16x16x32_f16(false, A[1], false, Bf[nt * 2 + 1],
                                                   (short)0, c, false, false);
        acc[nt] = c;
    }

    // --- scatter D: row = tile*16 + half*8 + r, col = nt*16 + m
#pragma unroll
    for (int nt = 0; nt < 4; ++nt) {
        int col = nt * 16 + m;
#pragma unroll
        for (int r = 0; r < 8; ++r) {
            int row = wave * 16 + half * 8 + r;
            if (col < 32) z1[(size_t)row * 32 + col]        = acc[nt][r];
            else          r1[(size_t)row * 32 + (col - 32)] = acc[nt][r];
        }
    }
}

// ---------------------------------------------------------------------------
// K3: layer-2 edge aggregation, 32 channels wide. One wave per edge:
// lane k does aggz[dst*32+k] += z1[src*32+k] (coalesced 128B gather + atomics,
// both arrays are L2-resident: 12.8 MB each vs 192 MB L2).
// ---------------------------------------------------------------------------
__global__ void edge_pass2(const int* __restrict__ src, const int* __restrict__ dst,
                           const float* __restrict__ z1, float* __restrict__ aggz,
                           int nEdges) {
    int wave = (int)((blockIdx.x * blockDim.x + threadIdx.x) >> 5);
    int lane = threadIdx.x & 31;
    if (wave >= nEdges) return;
    int s = src[wave];
    int d = dst[wave];
    float v = z1[(size_t)s * 32 + lane];
    atomicAdd(&aggz[(size_t)d * 32 + lane], v);
}

// ---------------------------------------------------------------------------
// K4: h2 = relu(aggz + r1 + b2); h3 = relu(h2 @ W3^T + b3) via WMMA;
// then scalar 16->8->1 tail and pooled atomics. One wave per 16-node tile.
// All waves in every block reach __syncthreads (grid sized to padded N).
// ---------------------------------------------------------------------------
__global__ void tail_pool(const float* __restrict__ aggz,
                          const float* __restrict__ r1,
                          const float* __restrict__ b2,
                          const float* __restrict__ W3, const float* __restrict__ b3,
                          const float* __restrict__ W4, const float* __restrict__ b4,
                          const float* __restrict__ W5, const float* __restrict__ b5,
                          const int* __restrict__ bidx,
                          float* __restrict__ sums, float* __restrict__ cnts,
                          int nNodes) {
    __shared__ float h3buf[8][16][17];       // padded stride vs LDS bank conflicts
    int wib  = threadIdx.x >> 5;
    int wave = blockIdx.x * (blockDim.x >> 5) + wib;
    int lane = threadIdx.x & 31;
    int m    = lane & 15;
    int half = lane >> 4;

    // B fragment for W3 (16 out x 32 in): B[k][n] = W3[n*32+k]
    v16h Bf;
#pragma unroll
    for (int idx = 0; idx < 16; ++idx)
        Bf[idx] = (_Float16)W3[m * 32 + half * 16 + idx];

    // A fragment: h2 row (wave*16+m); padded rows read zero-initialized ws.
    int nodeRow = wave * 16 + m;
    const float* az = aggz + (size_t)nodeRow * 32;
    const float* rr = r1   + (size_t)nodeRow * 32;
    v16h A;
#pragma unroll
    for (int idx = 0; idx < 16; ++idx) {
        int k = half * 16 + idx;
        float h = az[k] + rr[k] + b2[k];
        A[idx] = (_Float16)(h > 0.f ? h : 0.f);
    }

    v8f c = {};
    c = __builtin_amdgcn_wmma_f32_16x16x32_f16(false, A, false, Bf, (short)0, c, false, false);

    // D: lane holds column n=m for rows half*8+r -> transpose through LDS
#pragma unroll
    for (int r = 0; r < 8; ++r) {
        float v = c[r] + b3[m];
        h3buf[wib][half * 8 + r][m] = v > 0.f ? v : 0.f;
    }
    __syncthreads();

    if (lane < 16) {
        int node = wave * 16 + lane;
        if (node < nNodes) {
            float h3r[16];
#pragma unroll
            for (int j = 0; j < 16; ++j) h3r[j] = h3buf[wib][lane][j];
            float h4[8];
#pragma unroll
            for (int o = 0; o < 8; ++o) {
                float s = b4[o];
#pragma unroll
                for (int j = 0; j < 16; ++j) s = fmaf(W4[o * 16 + j], h3r[j], s);
                h4[o] = s > 0.f ? s : 0.f;
            }
            float s5 = b5[0];
#pragma unroll
            for (int o = 0; o < 8; ++o) s5 = fmaf(W5[o], h4[o], s5);
            int g = bidx[node];
            atomicAdd(&sums[g], s5);
            atomicAdd(&cnts[g], 1.0f);
        }
    }
}

// ---------------------------------------------------------------------------
// K5: pooled mean + sigmoid
// ---------------------------------------------------------------------------
__global__ void finalize_pool(const float* __restrict__ sums, const float* __restrict__ cnts,
                              float* __restrict__ out, int nGraphs) {
    int i = blockIdx.x * blockDim.x + threadIdx.x;
    if (i < nGraphs) {
        float cnt = cnts[i];
        cnt = cnt > 1.f ? cnt : 1.f;
        float p = sums[i] / cnt;
        out[i] = 1.f / (1.f + expf(-p));
    }
}

// ---------------------------------------------------------------------------
extern "C" void kernel_launch(void* const* d_in, const int* in_sizes, int n_in,
                              void* d_out, int out_size, void* d_ws, size_t ws_size,
                              hipStream_t stream) {
    const float* feat    = (const float*)d_in[0];
    const int*   graph   = (const int*)d_in[1];
    const int*   bidx    = (const int*)d_in[2];
    const float* W1_rel  = (const float*)d_in[4];
    const float* W1_root = (const float*)d_in[5];
    const float* b1      = (const float*)d_in[6];
    const float* W2_rel  = (const float*)d_in[7];
    const float* W2_root = (const float*)d_in[8];
    const float* b2      = (const float*)d_in[9];
    const float* W3      = (const float*)d_in[10];
    const float* b3      = (const float*)d_in[11];
    const float* W4      = (const float*)d_in[12];
    const float* b4      = (const float*)d_in[13];
    const float* W5      = (const float*)d_in[14];
    const float* b5      = (const float*)d_in[15];

    const int N = in_sizes[0];
    const int E = in_sizes[1] / 2;
    const int G = out_size;
    const int* src = graph;
    const int* dst = graph + E;

    // Tiling: 16 nodes per wave, 8 waves (256 threads) per block.
    const int nTiles  = (N + 15) / 16;                 // 6250
    const int blocksT = (nTiles * 32 + 255) / 256;     // 782
    const int NPAD    = blocksT * 8 * 16;              // 100096 (full grid coverage for K4)

    // Workspace layout (floats): agg1[NPAD] | z1[NPAD*32] | r1[NPAD*32] |
    //                            aggz[NPAD*32] | sums[G] | cnts[G]
    float* ws   = (float*)d_ws;
    float* agg1 = ws;
    float* z1   = agg1 + NPAD;
    float* r1   = z1   + (size_t)NPAD * 32;
    float* aggz = r1   + (size_t)NPAD * 32;
    float* sums = aggz + (size_t)NPAD * 32;
    float* cnts = sums + G;
    size_t usedBytes = ((size_t)NPAD * 97 + 2 * (size_t)G) * sizeof(float);

    hipMemsetAsync(d_ws, 0, usedBytes, stream);

    edge_pass1<<<(E + 255) / 256, 256, 0, stream>>>(src, dst, feat, agg1, E);

    node_transform<<<blocksT, 256, 0, stream>>>(feat, agg1, W1_rel, W1_root, b1,
                                                W2_rel, W2_root, z1, r1, nTiles);

    edge_pass2<<<(E + 7) / 8, 256, 0, stream>>>(src, dst, z1, aggz, E);

    tail_pool<<<blocksT, 256, 0, stream>>>(aggz, r1, b2, W3, b3, W4, b4, W5, b5,
                                           bidx, sums, cnts, N);

    finalize_pool<<<(G + 255) / 256, 256, 0, stream>>>(sums, cnts, (float*)d_out, G);
}
